// SynthesisLayer_32083405701190
// MI455X (gfx1250) — matrix operand order repaired and checked by reading, not compiled, not run on hardware
//
#include <hip/hip_runtime.h>

#define B_    16
#define CIN   512
#define COUT  512
#define RES   64
#define WDIM  512
#define KK    4608   // CIN * 9
#define LDA   40     // LDS row stride (ushorts) for A tile
#define LDB   40     // LDS row stride (ushorts) for B tile (transposed [n][k])
#define NSTEP 144    // KK / 32

typedef __attribute__((ext_vector_type(16))) __bf16 v16bf;
typedef __attribute__((ext_vector_type(8)))  float  v8f;

union Frag { v16bf v; unsigned int u[8]; };

__device__ __forceinline__ unsigned short f32_to_bf16(float f) {
  unsigned int u = __builtin_bit_cast(unsigned int, f);
  u += 0x7FFFu + ((u >> 16) & 1u);   // round-to-nearest-even
  return (unsigned short)(u >> 16);
}

// LDS byte offset of a pointer into a __shared__ array (generic -> AS(3) -> int)
__device__ __forceinline__ unsigned lds_addr32(const void* p) {
  return (unsigned)(unsigned long long)
      (__attribute__((address_space(3))) const void*)p;
}

// ---------------------------------------------------------------------------
// 1) styles[b][cin] = dot(w[b,:], affine_w[cin,:]) / sqrt(WDIM) + affine_b[cin]
// ---------------------------------------------------------------------------
__global__ __launch_bounds__(256) void styles_kernel(
    const float* __restrict__ w, const float* __restrict__ affine_w,
    const float* __restrict__ affine_b, float* __restrict__ styles) {
  __shared__ float wrow[WDIM];
  const int b   = blockIdx.y;
  const int cin = blockIdx.x * 256 + threadIdx.x;
  for (int i = threadIdx.x; i < WDIM; i += 256) wrow[i] = w[b * WDIM + i];
  __syncthreads();
  const float gain = 0.04419417382415922f;  // 1/sqrt(512)
  const float* ar = affine_w + (size_t)cin * WDIM;
  float acc = 0.f;
  #pragma unroll 4
  for (int d = 0; d < WDIM; ++d) acc += wrow[d] * ar[d];
  styles[b * CIN + cin] = acc * gain + affine_b[cin];
}

// ---------------------------------------------------------------------------
// 2) amat[b][cout][r*512+cin] = bf16( weight[cout][cin][r]/sqrt(KK) * style * demod )
// ---------------------------------------------------------------------------
__global__ __launch_bounds__(256) void modweight_kernel(
    const float* __restrict__ weight, const float* __restrict__ styles,
    unsigned short* __restrict__ amat) {
  const int cout = blockIdx.x;
  const int b    = blockIdx.y;
  const int tid  = threadIdx.x;
  __shared__ float ssty[CIN];
  __shared__ float red[256];
  for (int i = tid; i < CIN; i += 256) ssty[i] = styles[b * CIN + i];
  __syncthreads();
  const float wgain = 0.014731391274719739f;  // 1/sqrt(4608)
  const float* wrow = weight + (size_t)cout * KK;
  float vals[18];
  float ss = 0.f;
  #pragma unroll
  for (int j = 0; j < 18; ++j) {
    int idx = tid + j * 256;           // idx = cin*9 + r
    int cin = idx / 9;
    float v = wrow[idx] * wgain * ssty[cin];
    vals[j] = v;
    ss += v * v;
  }
  red[tid] = ss;
  __syncthreads();
  for (int off = 128; off > 0; off >>= 1) {
    if (tid < off) red[tid] += red[tid + off];
    __syncthreads();
  }
  const float dmod = rsqrtf(red[0] + 1e-8f);
  unsigned short* arow = amat + ((size_t)b * COUT + cout) * KK;
  #pragma unroll
  for (int j = 0; j < 18; ++j) {
    int idx = tid + j * 256;
    int cin = idx / 9;
    int r   = idx - cin * 9;
    arow[r * CIN + cin] = f32_to_bf16(vals[j] * dmod);
  }
}

// ---------------------------------------------------------------------------
// 2b) one-pass streaming x (f32) -> xb (bf16)
// ---------------------------------------------------------------------------
__global__ __launch_bounds__(256) void xcast_kernel(
    const float* __restrict__ x, unsigned short* __restrict__ xb) {
  size_t i = ((size_t)blockIdx.x * 256 + threadIdx.x) * 8;
  float4 a, b;
  __builtin_memcpy(&a, x + i, 16);
  __builtin_memcpy(&b, x + i + 4, 16);
  unsigned short t[8] = {
    f32_to_bf16(a.x), f32_to_bf16(a.y), f32_to_bf16(a.z), f32_to_bf16(a.w),
    f32_to_bf16(b.x), f32_to_bf16(b.y), f32_to_bf16(b.z), f32_to_bf16(b.w)};
  __builtin_memcpy(xb + i, t, 16);
}

// ---------------------------------------------------------------------------
// 3) implicit-GEMM conv: per workgroup, 128 cout x 128 pixels (two image rows),
//    K = 4608 in 144 steps of 32.  Each wave: 2 A-frags x 4 B-frags -> 8 WMMA
//    per step (A operand reused 6/8).  Double-buffered LDS; A tile staged with
//    async global->LDS (ASYNCcnt).
// ---------------------------------------------------------------------------
__global__ __launch_bounds__(256) void conv_kernel(
    const unsigned short* __restrict__ amat,
    const unsigned short* __restrict__ xb,
    const float* __restrict__ noise,
    const float* __restrict__ bias,
    const float* __restrict__ nstrength,
    float* __restrict__ out) {
  __shared__ unsigned short As[2][128 * LDA];   // [m=cout][k]
  __shared__ unsigned short Bs[2][128 * LDB];   // [n=pixel][k]  (transposed)

  const int tid   = threadIdx.x;
  const int bz    = blockIdx.z;            // batch
  const int h0    = blockIdx.y * 2;        // first of two image rows
  const int cout0 = blockIdx.x * 128;

  const int lane = tid & 31;
  const int wave = tid >> 5;               // 8 waves
  const int mrow = wave >> 1;              // M subtile pair 0..3 -> subtiles 2*mrow, 2*mrow+1
  const int row  = wave & 1;               // image row within tile
  const int ng   = row * 4;                // first of 4 N subtiles
  const int lm   = lane & 15;
  const int lh   = lane >> 4;

  v8f acc00 = {}, acc01 = {}, acc02 = {}, acc03 = {};
  v8f acc10 = {}, acc11 = {}, acc12 = {}, acc13 = {};

  // A staging: 128 rows x 32 bf16, each thread two 16B async chunks
  const int arow = tid >> 2;               // 0..63 (second chunk at +64)
  const int acol = (tid & 3) << 3;
  const unsigned short* abase =
      amat + ((size_t)bz * COUT + cout0 + arow) * KK + acol;

  // B staging: thread -> (cin pair bk0,bk0+1) x (8 pixel cols), packed b32 stores
  const int bk0 = (tid >> 4) << 1;         // 0,2,..,30
  const int bn0 = (tid & 15) << 3;         // pixel base 0..120 (stays in one row)

  auto stageA = [&](int s, int buf) {
    const int rr   = s >> 4;
    const int cin0 = (s & 15) << 5;
    const unsigned short* ga0 = abase + rr * 512 + cin0;
    const unsigned short* ga1 = ga0 + (size_t)64 * KK;          // rows arow+64
    unsigned l0 = lds_addr32(&As[buf][arow * LDA + acol]);
    unsigned l1 = l0 + 64 * LDA * 2;
    asm volatile("global_load_async_to_lds_b128 %0, %1, off"
                 :: "v"(l0), "v"(ga0) : "memory");
    asm volatile("global_load_async_to_lds_b128 %0, %1, off"
                 :: "v"(l1), "v"(ga1) : "memory");
  };

  auto stageB = [&](int s, int buf) {
    const int rr   = s >> 4;
    const int cin0 = (s & 15) << 5;
    const int dh   = rr / 3 - 1;
    const int dw   = rr % 3 - 1;
    const int nrow = bn0 >> 6;             // which of the two image rows
    const int wb   = bn0 & 63;
    const int hh   = h0 + nrow + dh;
    unsigned short t0[8], t1[8];
    if ((unsigned)hh < (unsigned)RES) {
      const unsigned short* s0 =
          xb + (((size_t)bz * CIN + (cin0 + bk0)) * RES + hh) * RES;
      const unsigned short* s1 = s0 + RES * RES;   // cin + 1
      const int w0 = wb + dw;
      if (w0 >= 0 && w0 + 7 < RES) {
        __builtin_memcpy(t0, s0 + w0, 16);
        __builtin_memcpy(t1, s1 + w0, 16);
      } else {
        #pragma unroll
        for (int i = 0; i < 8; ++i) {
          int ww = wb + i + dw;
          bool in = (ww >= 0 && ww < RES);
          t0[i] = in ? s0[ww] : (unsigned short)0;
          t1[i] = in ? s1[ww] : (unsigned short)0;
        }
      }
    } else {
      #pragma unroll
      for (int i = 0; i < 8; ++i) { t0[i] = 0; t1[i] = 0; }
    }
    #pragma unroll
    for (int i = 0; i < 8; ++i) {
      unsigned pk = (unsigned)t0[i] | ((unsigned)t1[i] << 16);
      *(unsigned*)(&Bs[buf][(bn0 + i) * LDB + bk0]) = pk;
    }
  };

  // prologue: stage tile 0
  stageA(0, 0);
  stageB(0, 0);
  asm volatile("s_wait_asynccnt 0" ::: "memory");
  __syncthreads();

  for (int s = 0; s < NSTEP; ++s) {
    const int cur = s & 1;
    const int nxt = cur ^ 1;

    // fire next A tile early (async, no VGPR round-trip)
    if (s + 1 < NSTEP) stageA(s + 1, nxt);

    // ---- fragments per ISA 7.12.2 VGPR layouts (from buffer `cur`) ----
    Frag a0, a1, b0, b1, b2, b3;
    #pragma unroll
    for (int v = 0; v < 8; ++v) {          // A 16x32: k = (v>=4?16:0)+8*lh+2*(v&3)
      int k = ((v >> 2) << 4) + (lh << 3) + ((v & 3) << 1);
      a0.u[v] = *(const unsigned int*)(&As[cur][((mrow * 2 + 0) * 16 + lm) * LDA + k]);
      a1.u[v] = *(const unsigned int*)(&As[cur][((mrow * 2 + 1) * 16 + lm) * LDA + k]);
    }
    #pragma unroll
    for (int v = 0; v < 8; ++v) {          // B 32x16: k = (v>=4?16:0)+8*lh+2*(v&3), n = lm
      int k = ((v >> 2) << 4) + (lh << 3) + ((v & 3) << 1);
      b0.u[v] = *(const unsigned int*)(&Bs[cur][((ng + 0) * 16 + lm) * LDB + k]);
      b1.u[v] = *(const unsigned int*)(&Bs[cur][((ng + 1) * 16 + lm) * LDB + k]);
      b2.u[v] = *(const unsigned int*)(&Bs[cur][((ng + 2) * 16 + lm) * LDB + k]);
      b3.u[v] = *(const unsigned int*)(&Bs[cur][((ng + 3) * 16 + lm) * LDB + k]);
    }

    // stage next B tile into the other buffer (overlaps with WMMA below)
    if (s + 1 < NSTEP) stageB(s + 1, nxt);

    acc00 = __builtin_amdgcn_wmma_f32_16x16x32_bf16(
        false, a0.v, false, b0.v, (short)0, acc00, false, false);
    acc01 = __builtin_amdgcn_wmma_f32_16x16x32_bf16(
        false, a0.v, false, b1.v, (short)0, acc01, true, false);  // reuse A
    acc02 = __builtin_amdgcn_wmma_f32_16x16x32_bf16(
        false, a0.v, false, b2.v, (short)0, acc02, true, false);  // reuse A
    acc03 = __builtin_amdgcn_wmma_f32_16x16x32_bf16(
        false, a0.v, false, b3.v, (short)0, acc03, true, false);  // reuse A
    acc10 = __builtin_amdgcn_wmma_f32_16x16x32_bf16(
        false, a1.v, false, b0.v, (short)0, acc10, false, false);
    acc11 = __builtin_amdgcn_wmma_f32_16x16x32_bf16(
        false, a1.v, false, b1.v, (short)0, acc11, true, false);  // reuse A
    acc12 = __builtin_amdgcn_wmma_f32_16x16x32_bf16(
        false, a1.v, false, b2.v, (short)0, acc12, true, false);  // reuse A
    acc13 = __builtin_amdgcn_wmma_f32_16x16x32_bf16(
        false, a1.v, false, b3.v, (short)0, acc13, true, false);  // reuse A

    asm volatile("s_wait_asynccnt 0" ::: "memory");
    __syncthreads();
  }

  // ---- fused epilogue: noise + bias + lrelu*sqrt(2) + clamp ----
  const int hrow = h0 + row;
  const float ns = nstrength[0];
  const float* nrow = noise + ((size_t)bz * RES + hrow) * RES;
  v8f accs[2][4] = {{acc00, acc01, acc02, acc03}, {acc10, acc11, acc12, acc13}};
  #pragma unroll
  for (int mt = 0; mt < 2; ++mt) {
    #pragma unroll
    for (int t = 0; t < 4; ++t) {
      const int wpix = t * 16 + lm;        // column within image row
      const float nz = nrow[wpix] * ns;
      #pragma unroll
      for (int i = 0; i < 8; ++i) {
        const int m    = (mrow * 2 + mt) * 16 + (lh << 3) + i;  // C/D: M = i + 8*lh
        const int cout = cout0 + m;
        float v = accs[mt][t][i] + nz + bias[cout];
        v = (v >= 0.f ? v : 0.2f * v) * 1.4142135623730951f;
        v = fminf(fmaxf(v, -256.f), 256.f);
        out[(((size_t)bz * COUT + cout) * RES + hrow) * RES + wpix] = v;
      }
    }
  }
}

// ---------------------------------------------------------------------------
extern "C" void kernel_launch(void* const* d_in, const int* in_sizes, int n_in,
                              void* d_out, int out_size, void* d_ws, size_t ws_size,
                              hipStream_t stream) {
  (void)in_sizes; (void)n_in; (void)out_size; (void)ws_size;
  const float* x        = (const float*)d_in[0];
  const float* w        = (const float*)d_in[1];
  const float* weight   = (const float*)d_in[2];
  const float* affine_w = (const float*)d_in[3];
  const float* affine_b = (const float*)d_in[4];
  const float* bias     = (const float*)d_in[5];
  const float* noise    = (const float*)d_in[6];
  const float* nstr     = (const float*)d_in[7];
  float* out = (float*)d_out;

  // workspace layout: styles (32 KB) | amat (75.5 MB) | xb (67 MB)
  float*          styles = (float*)d_ws;
  unsigned short* amat   = (unsigned short*)((char*)d_ws + 64 * 1024);
  unsigned short* xb     = (unsigned short*)((char*)d_ws + 64 * 1024 +
                                             (size_t)B_ * COUT * KK * 2);

  styles_kernel<<<dim3(CIN / 256, B_), 256, 0, stream>>>(w, affine_w, affine_b, styles);
  modweight_kernel<<<dim3(COUT, B_), 256, 0, stream>>>(weight, styles, amat);
  xcast_kernel<<<dim3((B_ * CIN * RES * RES) / 2048), 256, 0, stream>>>(x, xb);
  conv_kernel<<<dim3(COUT / 128, RES / 2, B_), 256, 0, stream>>>(amat, xb, noise, bias, nstr, out);
}
